// LiquidSeparator_49495203119913
// MI455X (gfx1250) — matrix operand order repaired and checked by reading, not compiled, not run on hardware
//
#include <hip/hip_runtime.h>
#include <hip/hip_bf16.h>

// ---------------------------------------------------------------------------
// CfC liquid-network stack for MI455X (gfx1250).
//  - bf16 WMMA (v_wmma_f32_16x16x32_bf16) for all cell GEMMs, f32 accumulate
//  - weights pre-packed to WMMA B-fragment layout (bf16), streamed from L2
//  - hidden state + xcat staged in LDS (bf16); rows m>=4 of the A matrix are
//    served by one shared zero row (no predication, no per-step re-zeroing)
//  - double-buffered k-loop (prefetch distance 1), fully unrolled per cell
//  - 4 layers pipelined across 4 WGPs with acquire/release progress counters
// ---------------------------------------------------------------------------

typedef __attribute__((ext_vector_type(16))) __bf16 v16bf;
typedef __attribute__((ext_vector_type(8)))  float  v8f;

struct U128 { unsigned a, b, c, d; };          // POD 16-byte chunk
union  Frag { v16bf v; U128 q[2]; };           // one WMMA operand per lane

__device__ __forceinline__ unsigned short f2bf(float f) {
    unsigned u = __float_as_uint(f);
    return (unsigned short)((u + 0x7FFFu + ((u >> 16) & 1u)) >> 16);  // RNE
}
__device__ __forceinline__ float bf2f(unsigned short h) {
    return __uint_as_float(((unsigned)h) << 16);
}

// ---------------------------------------------------------------------------
// Weight packing: fuse [W_ff1|W_ff2|W_ta|W_tb] into one bf16 buffer in
// B-fragment order. Fragment layout for 16x16x32 bf16 B (32(K) x 16(N)):
//   lanes 0-15 : n = lane,     k = kbase + (j&7) + 16*(j>>3)      (j = 0..15)
//   lanes 16-31: n = lane-16,  k = kbase + 8 + (j&7) + 16*(j>>3)
// dst element index = ((ntile*Kt + ktile)*32 + lane)*16 + j
// Rows outside the real K extent (alignment padding) pack as zero so the
// padded xcat slices contribute nothing.
// ---------------------------------------------------------------------------
__global__ void pack_w_kernel(const float* __restrict__ W1, const float* __restrict__ W2,
                              const float* __restrict__ Wa, const float* __restrict__ Wb,
                              unsigned short* __restrict__ dst,
                              int H, int HP, int seg0r, int seg0p, int Kpad, int NF)
{
    size_t total = (size_t)NF * (size_t)Kpad;
    size_t idx   = (size_t)blockIdx.x * 256u + threadIdx.x;
    if (idx >= total) return;
    int Kt   = Kpad >> 5;
    int f    = (int)(idx >> 9);        // fragment id = ntile*Kt + ktile
    int r    = (int)(idx & 511);
    int lane = r >> 4, j = r & 15;
    int nt   = f / Kt, kt = f % Kt;
    int n = nt * 16 + (lane & 15);
    int k = kt * 32 + ((lane >> 4) << 3) + (j & 7) + ((j >> 3) << 4);
    int q = n / HP, u = n % HP;
    float val = 0.f;
    if (u < H) {
        int row = -1;
        if (k < seg0r)                         row = k;                    // input part
        else if (k >= seg0p && k < seg0p + H)  row = seg0r + (k - seg0p);  // hidden part
        if (row >= 0) {
            const float* W = (q == 0) ? W1 : (q == 1) ? W2 : (q == 2) ? Wa : Wb;
            val = W[(size_t)row * H + u];
        }
    }
    dst[idx] = f2bf(val);
}

__global__ void pack_b_kernel(const float* __restrict__ b1, const float* __restrict__ b2,
                              const float* __restrict__ ba, const float* __restrict__ bb,
                              float* __restrict__ dst, int H, int HP)
{
    int i = blockIdx.x * 256 + threadIdx.x;
    if (i >= 4 * HP) return;
    int q = i / HP, u = i % HP;
    float v = 0.f;
    if (u < H) {
        const float* b = (q == 0) ? b1 : (q == 1) ? b2 : (q == 2) ? ba : bb;
        v = b[u];
    }
    dst[i] = v;
}

// ---------------------------------------------------------------------------
// Per-cell fused GEMM: raw[m][q*HP+u] = xcat @ [W1|W2|Wa|Wb] + bias
// A (16x32 bf16) rows 0..3 come from LDS xcat; rows 4..15 read the shared
// zero row (index 4). Double-buffered, fully unrolled k-loop.
// ---------------------------------------------------------------------------
template <int KT>
__device__ __forceinline__ void cell_gemm(const unsigned short* __restrict__ sx,
                                          int xstr, int slice, int nt,
                                          const U128* __restrict__ wp,
                                          const float* __restrict__ fb,
                                          float* __restrict__ raw, int tid)
{
    const int wv = tid >> 5, lane = tid & 31;
    const int m  = lane & 15;
    const int row = (m < 4) ? m : 4;                // rows 4..15 -> zero row
    const int khalf = (lane >> 4) << 3;             // 0 or 8 elements
    const unsigned short* arow = sx + row * xstr + slice + khalf;
    for (int tile = wv; tile < nt; tile += 16) {
        const U128* wt = wp + (size_t)tile * (KT * 64) + lane * 2;
        v8f acc = {0.f, 0.f, 0.f, 0.f, 0.f, 0.f, 0.f, 0.f};
        Frag a0, b0, a1, b1;
        {   // prologue: k-tile 0
            const U128* ap = (const U128*)arow;
            a0.q[0] = ap[0]; a0.q[1] = ap[2];
            b0.q[0] = wt[0]; b0.q[1] = wt[1];
        }
#pragma unroll
        for (int kt = 0; kt < KT; ++kt) {
            if ((kt & 1) == 0) {
                if (kt + 1 < KT) {                  // prefetch k-tile kt+1
                    const U128* ap = (const U128*)(arow + (kt + 1) * 32);
                    a1.q[0] = ap[0]; a1.q[1] = ap[2];
                    b1.q[0] = wt[(kt + 1) * 64]; b1.q[1] = wt[(kt + 1) * 64 + 1];
                }
                acc = __builtin_amdgcn_wmma_f32_16x16x32_bf16(
                          false, a0.v, false, b0.v, (short)0, acc, false, false);
            } else {
                if (kt + 1 < KT) {
                    const U128* ap = (const U128*)(arow + (kt + 1) * 32);
                    a0.q[0] = ap[0]; a0.q[1] = ap[2];
                    b0.q[0] = wt[(kt + 1) * 64]; b0.q[1] = wt[(kt + 1) * 64 + 1];
                }
                acc = __builtin_amdgcn_wmma_f32_16x16x32_bf16(
                          false, a1.v, false, b1.v, (short)0, acc, false, false);
            }
        }
        if (lane < 16) {                 // D rows m=0..3 live in VGPR 0..3, lanes 0..15
            int col = tile * 16 + lane;
            float bias = fb[col];
            raw[0 * 2048 + col] = acc[0] + bias;
            raw[1 * 2048 + col] = acc[1] + bias;
            raw[2 * 2048 + col] = acc[2] + bias;
            raw[3 * 2048 + col] = acc[3] + bias;
        }
    }
}

// CfC gating: h = tanh(z1)*(1-s) + s*tanh(z2),  s = sigmoid(za+zb)
__device__ __forceinline__ void cell_combine(unsigned short* __restrict__ sx,
                                             int xstr, int seg, int H, int HP,
                                             const float* __restrict__ raw, int tid)
{
    for (int i = tid; i < 4 * HP; i += 512) {
        int m = i & 3, u = i >> 2;
        unsigned short out = 0;
        if (u < H) {
            float z1 = raw[m * 2048 + u];
            float z2 = raw[m * 2048 + HP + u];
            float za = raw[m * 2048 + 2 * HP + u];
            float zb = raw[m * 2048 + 3 * HP + u];
            float s  = 1.f / (1.f + __expf(-(za + zb)));
            float v  = tanhf(z1) * (1.f - s) + s * tanhf(z2);
            out = f2bf(v);
        }
        sx[m * xstr + seg + u] = out;     // padded units stay zero
    }
}

struct KParams {
    const float* x;                 // [4,256,4000]
    float*       out;               // [4,512,4000]
    const unsigned short* wpack;    // packed bf16 weights, all layers
    const float* bias;              // packed fused biases, all layers
    unsigned short* seq0;           // bf16 [T,4,512] layer0 -> layer1
    unsigned short* seq1;
    unsigned short* seq2;
    unsigned*    prog;              // per-layer progress counters
    const float* gamma[4];
    const float* beta[4];
};

// Per-layer body, specialized on IN so every k-loop trip count is constant.
template <int IN>
__device__ __forceinline__ void run_layer(const KParams& p, int L, int tid,
                                          unsigned short* s_x, float* s_raw,
                                          float* s_ln)
{
    constexpr int XSTR = IN + 800;       // bf16 row stride of LDS xcat buffer
    constexpr int OFF2 = IN + 160;       // h2 segment start (h1 padded 154->160)
    constexpr int OFF3 = IN + 272;       // h3 segment start (h2 padded 102->112)
    constexpr int KT1  = (IN + 160) / 32;

    // Layer-local packed-weight / bias pointers
    const size_t szc2 = (size_t)448 * 288, szc3 = (size_t)2048 * 640;
    const size_t wl0  = (size_t)640 * 416 + szc2 + szc3;
    const size_t wl   = (size_t)640 * 672 + szc2 + szc3;
    const unsigned short* w_l = p.wpack + ((L == 0) ? 0 : wl0 + (size_t)(L - 1) * wl);
    const size_t c1sz = (size_t)640 * (IN + 160);
    const U128* wc0 = (const U128*)w_l;
    const U128* wc1 = (const U128*)(w_l + c1sz);
    const U128* wc2 = (const U128*)(w_l + c1sz + szc2);
    const float* fb = p.bias + L * 3136;

    const unsigned short* inseq  = (L == 1) ? p.seq0 : (L == 2) ? p.seq1 : p.seq2;
    unsigned short*       outseq = (L == 0) ? p.seq0 : (L == 1) ? p.seq1 : p.seq2;
    const float* gamma = p.gamma[L];
    const float* beta  = p.beta[L];

    for (int t = 0; t < 4000; ++t) {
        if (L > 0) {                                  // wait for producer layer
            if (tid == 0) {
                while (__hip_atomic_load(&p.prog[L - 1], __ATOMIC_ACQUIRE,
                                         __HIP_MEMORY_SCOPE_AGENT) <= (unsigned)t)
                    __builtin_amdgcn_s_sleep(1);
            }
            __syncthreads();
        }
        // stage x_t (bf16) into LDS rows 0..3
        for (int i = tid; i < 4 * IN; i += 512) {
            int c = i % IN, m = i / IN;
            float v = (L == 0) ? p.x[((size_t)m * 256 + c) * 4000 + t]
                               : bf2f(inseq[(size_t)t * 2048 + m * 512 + c]);
            s_x[m * XSTR + c] = f2bf(v);
        }
        __syncthreads();

        // cell 1: xcat=[x|h1], K=IN+160, N=4*160
        cell_gemm<KT1>(s_x, XSTR, 0, 40, wc0, fb, s_raw, tid);
        __syncthreads();
        cell_combine(s_x, XSTR, IN, 154, 160, s_raw, tid);
        __syncthreads();
        // cell 2: xcat=[h1|h2], K=288, N=4*112
        cell_gemm<9>(s_x, XSTR, IN, 28, wc1, fb + 640, s_raw, tid);
        __syncthreads();
        cell_combine(s_x, XSTR, OFF2, 102, 112, s_raw, tid);
        __syncthreads();
        // cell 3: xcat=[h2|h3], K=640, N=4*512
        cell_gemm<20>(s_x, XSTR, OFF2, 128, wc2, fb + 1088, s_raw, tid);
        __syncthreads();
        cell_combine(s_x, XSTR, OFF3, 512, 512, s_raw, tid);
        __syncthreads();

        // LayerNorm stats over h3 (4 waves, one batch row each)
        if (tid < 128) {
            int m = tid >> 5, ln = tid & 31;
            float s = 0.f, ss = 0.f;
            for (int j = ln; j < 512; j += 32) {
                float v = bf2f(s_x[m * XSTR + OFF3 + j]);
                s += v; ss += v * v;
            }
            for (int d = 16; d > 0; d >>= 1) {
                s  += __shfl_xor(s,  d, 32);
                ss += __shfl_xor(ss, d, 32);
            }
            if (ln == 0) {
                float mu  = s * (1.f / 512.f);
                float var = ss * (1.f / 512.f) - mu * mu;
                s_ln[m] = mu; s_ln[4 + m] = rsqrtf(var + 1e-5f);
            }
        }
        __syncthreads();

        // LN + residual + output
        for (int i = tid; i < 2048; i += 512) {
            int n = i & 511, m = i >> 9;
            float v = bf2f(s_x[m * XSTR + OFF3 + n]);
            float y = (v - s_ln[m]) * s_ln[4 + m] * gamma[n] + beta[n];
            if (L > 0) y += bf2f(inseq[(size_t)t * 2048 + m * 512 + n]);
            if (L == 3) p.out[((size_t)m * 512 + n) * 4000 + t] = y;   // [B,OUT,T]
            else        outseq[(size_t)t * 2048 + m * 512 + n] = f2bf(y);
        }
        __threadfence();
        __syncthreads();
        if (L < 3 && tid == 0)
            __hip_atomic_store(&p.prog[L], (unsigned)(t + 1),
                               __ATOMIC_RELEASE, __HIP_MEMORY_SCOPE_AGENT);
        __syncthreads();
    }
}

__global__ __launch_bounds__(512) void cfc_pipeline_kernel(KParams p)
{
    const int L   = blockIdx.x;     // one layer per block (per WGP)
    const int tid = threadIdx.x;

    // 5 rows: 4 real batch rows + 1 shared zero row for A lanes m>=4.
    __shared__ __align__(16) unsigned short s_x[5 * 1312];
    __shared__ float s_raw[4 * 2048];                        // pre-activation stash
    __shared__ float s_ln[8];                                // mu[4], rstd[4]

    for (int i = tid; i < 5 * 1312; i += 512) s_x[i] = 0;    // zero state + pads
    if (tid < 8) s_ln[tid] = 0.f;
    __syncthreads();

    if (L == 0) run_layer<256>(p, 0, tid, s_x, s_raw, s_ln);
    else        run_layer<512>(p, L, tid, s_x, s_raw, s_ln);
}

// ---------------------------------------------------------------------------
extern "C" void kernel_launch(void* const* d_in, const int* in_sizes, int n_in,
                              void* d_out, int out_size, void* d_ws, size_t ws_size,
                              hipStream_t stream)
{
    (void)n_in; (void)out_size; (void)ws_size;
    // Flatten-order detection: insertion order (x first) vs tree_leaves (alpha).
    const bool x_first = (in_sizes[0] == 4 * 256 * 4000);
    auto pW = [&](int l, int c, int q) {      // q: 0..3 -> W_ff1,W_ff2,W_ta,W_tb
        if (x_first) { static const int mp[4] = {0, 2, 4, 6};
                       return 1 + l * 26 + c * 8 + mp[q]; }
        return l * 26 + c * 8 + q;
    };
    auto pB = [&](int l, int c, int q) {      // q: 0..3 -> b_ff1,b_ff2,b_ta,b_tb
        if (x_first) { static const int mp[4] = {1, 3, 5, 7};
                       return 1 + l * 26 + c * 8 + mp[q]; }
        return l * 26 + c * 8 + 4 + q;
    };
    const int xi = x_first ? 0 : 104;
    auto gi = [&](int l) { return x_first ? 1 + l * 26 + 24 : l * 26 + 25; };
    auto bi = [&](int l) { return x_first ? 1 + l * 26 + 25 : l * 26 + 24; };

    // Workspace layout
    char* ws = (char*)d_ws;
    unsigned* prog   = (unsigned*)ws;                       // 256 B (flags)
    float* biasbuf   = (float*)(ws + 256);                  // 4*3136 f32
    unsigned short* wpack = (unsigned short*)(ws + 256 + 4 * 3136 * 4);
    const size_t wtot = ((size_t)640 * 416 + 448 * 288 + 2048 * 640)
                      + 3 * ((size_t)640 * 672 + 448 * 288 + 2048 * 640);
    size_t seq_off = (256 + 4 * 3136 * 4 + wtot * 2 + 255) & ~(size_t)255;
    unsigned short* seq0 = (unsigned short*)(ws + seq_off);
    unsigned short* seq1 = seq0 + (size_t)4000 * 2048;
    unsigned short* seq2 = seq1 + (size_t)4000 * 2048;

    hipMemsetAsync(ws, 0, 256, stream);                     // progress counters

    // Pack weights + biases (re-done each call; deterministic, L2-resident after)
    size_t woff = 0;
    for (int l = 0; l < 4; ++l) {
        const int IN = l ? 512 : 256;
        struct CD { int H, HP, s0r, s0p, Kpad, NF; };
        CD cd[3] = { {154, 160, IN,  IN,  IN + 160, 640},
                     {102, 112, 154, 160, 288,      448},
                     {512, 512, 102, 112, 640,      2048} };
        const int boff[3] = {0, 640, 1088};
        for (int c = 0; c < 3; ++c) {
            const CD& d = cd[c];
            size_t total = (size_t)d.NF * d.Kpad;
            int blocks = (int)((total + 255) / 256);
            pack_w_kernel<<<blocks, 256, 0, stream>>>(
                (const float*)d_in[pW(l, c, 0)], (const float*)d_in[pW(l, c, 1)],
                (const float*)d_in[pW(l, c, 2)], (const float*)d_in[pW(l, c, 3)],
                wpack + woff, d.H, d.HP, d.s0r, d.s0p, d.Kpad, d.NF);
            int bblocks = (4 * d.HP + 255) / 256;
            pack_b_kernel<<<bblocks, 256, 0, stream>>>(
                (const float*)d_in[pB(l, c, 0)], (const float*)d_in[pB(l, c, 1)],
                (const float*)d_in[pB(l, c, 2)], (const float*)d_in[pB(l, c, 3)],
                biasbuf + l * 3136 + boff[c], d.H, d.HP);
            woff += total;
        }
    }

    KParams p;
    p.x     = (const float*)d_in[xi];
    p.out   = (float*)d_out;
    p.wpack = wpack;
    p.bias  = biasbuf;
    p.seq0  = seq0; p.seq1 = seq1; p.seq2 = seq2;
    p.prog  = prog;
    for (int l = 0; l < 4; ++l) {
        p.gamma[l] = (const float*)d_in[gi(l)];
        p.beta[l]  = (const float*)d_in[bi(l)];
    }
    cfc_pipeline_kernel<<<4, 512, 0, stream>>>(p);
}
